// Attention_60421599920730
// MI455X (gfx1250) — matrix operand order repaired and checked by reading, not compiled, not run on hardware
//
#include <hip/hip_runtime.h>
#include <hip/hip_bf16.h>
#include <math.h>

// Problem sizes (fixed by the reference)
#define BB 16
#define SS 2048
#define HH 1024
#define GG 1024

typedef __bf16 bf16_t;
typedef bf16_t v16bf __attribute__((ext_vector_type(16)));
typedef bf16_t v8bf  __attribute__((ext_vector_type(8)));
typedef bf16_t v4bf  __attribute__((ext_vector_type(4)));
typedef float  v8f   __attribute__((ext_vector_type(8)));

// Native CDNA5 v_tanh_f32 if the toolchain exposes it.
#if __has_builtin(__builtin_amdgcn_tanhf)
#define TANHF(x) __builtin_amdgcn_tanhf(x)
#else
#define TANHF(x) tanhf(x)
#endif

// LDS A-tile row stride in elements: 1024 + 8 (16B pad) to break the
// row-stride-2048B 16-way bank collision on ds_load_b128 A-fragment reads.
#define LDA 1032
#define STILES 4                      // 64 encoder rows staged per block
#define SMEM_BYTES (STILES * 16 * LDA * sizeof(bf16_t) + 64 * sizeof(float))

// ---------------------------------------------------------------------------
// Kernel 1: hb[b][g] = hidden[b,:] . W[g, 0:1024] + bias[g]   (tiny GEMM)
// ---------------------------------------------------------------------------
__global__ void prep_hb_kernel(const float* __restrict__ hidden,
                               const float* __restrict__ W,
                               const float* __restrict__ bias,
                               float* __restrict__ hb) {
    const int t = threadIdx.x;
    const int b = t & 15;
    const int g = blockIdx.x * 16 + (t >> 4);
    const float* hrow = hidden + b * HH;
    const float* wrow = W + (size_t)g * (2 * HH);   // W_h row g (K-contiguous)
    float acc = 0.0f;
#pragma unroll 4
    for (int k = 0; k < HH; k += 4) {
        float4 h4 = *(const float4*)(hrow + k);
        float4 w4 = *(const float4*)(wrow + k);
        acc += h4.x * w4.x + h4.y * w4.y + h4.z * w4.z + h4.w * w4.w;
    }
    hb[b * GG + g] = acc + bias[g];
}

// ---------------------------------------------------------------------------
// Kernel 2: fused e_proj GEMM (bf16 WMMA, f32 acc) + tanh + dot(v) -> logits
// grid = (S/64, B), block = 256 threads (8 waves), 132 KB dynamic LDS.
// 132 KB LDS caps occupancy at 1 workgroup/WGP anyway, so tell the compiler
// (256, 1): single-wave-per-SIMD minimum -> large VGPR budget, no spills.
// ---------------------------------------------------------------------------
__global__ void __launch_bounds__(256, 1)
attn_logits_kernel(const float* __restrict__ enc,
                   const float* __restrict__ W,
                   const float* __restrict__ hb,
                   const float* __restrict__ v,
                   float* __restrict__ logits) {
    extern __shared__ __align__(16) char smem_raw[];
    bf16_t* sA = (bf16_t*)smem_raw;                       // [64][LDA] bf16
    float*  sLog = (float*)(smem_raw + STILES * 16 * LDA * sizeof(bf16_t));

    const int t    = threadIdx.x;
    const int lane = t & 31;
    const int wave = t >> 5;
    const int hh   = lane >> 4;      // lane half (0: lanes 0-15, 1: 16-31)
    const int lr   = lane & 15;

    const int b  = blockIdx.y;
    const int s0 = blockIdx.x * (STILES * 16);

    if (t < 64) sLog[t] = 0.0f;

    // ---- Stage 64 x 1024 f32 encoder rows into LDS as bf16 (row-major) ----
    const float* encB = enc + ((size_t)b * SS + s0) * HH;
#pragma unroll 8
    for (int i = 0; i < 64; ++i) {                 // 16384 float4 / 256 thr
        const int idx = t + i * 256;
        const int row = idx >> 8;                  // / 256 float4-per-row
        const int c4  = idx & 255;
        float4 e = *(const float4*)(encB + row * HH + c4 * 4);
        v4bf p;
        p[0] = (bf16_t)e.x; p[1] = (bf16_t)e.y;
        p[2] = (bf16_t)e.z; p[3] = (bf16_t)e.w;
        *(v4bf*)&sA[row * LDA + c4 * 4] = p;
    }
    __syncthreads();

    // ---- Main loop: 8 g-tiles per wave, K = 1024 in 32 steps of 32 ----
    for (int i = 0; i < 8; ++i) {
        const int nt = wave + i * 8;               // g-tile index 0..63
        const int g  = nt * 16 + lr;               // this lane's column
        // W_e row g, K-contiguous (row stride 2H), offset +H selects W_e.
        const float* wrow = W + (size_t)g * (2 * HH) + HH;

        v8f acc[STILES];
#pragma unroll
        for (int st = 0; st < STILES; ++st) acc[st] = (v8f){};

#pragma unroll 2
        for (int ks = 0; ks < 32; ++ks) {
            // B fragment: lane = column g, 16 consecutive K values
            // (lanes 0-15: K 0-15 of step, lanes 16-31: K 16-31).
            const float4* bp = (const float4*)(wrow + ks * 32 + hh * 16);
            __builtin_prefetch(wrow + (ks + 2) * 32 + hh * 16, 0, 3);
            union { v16bf vv; bf16_t e[16]; } bf;
#pragma unroll
            for (int j = 0; j < 4; ++j) {
                float4 q = bp[j];
                bf.e[4 * j + 0] = (bf16_t)q.x;
                bf.e[4 * j + 1] = (bf16_t)q.y;
                bf.e[4 * j + 2] = (bf16_t)q.z;
                bf.e[4 * j + 3] = (bf16_t)q.w;
            }
            // A fragments from LDS: lane row = lr (per s-tile), chunks
            // K = ks*32 + hh*16 .. +7  (V0-3)  and  +8 .. +15 (V4-7).
#pragma unroll
            for (int st = 0; st < STILES; ++st) {
                const v8bf* ap =
                    (const v8bf*)&sA[(st * 16 + lr) * LDA + ks * 32 + hh * 16];
                union { v16bf vv; v8bf h[2]; } af;
                af.h[0] = ap[0];
                af.h[1] = ap[1];                   // +8 elements
                acc[st] = __builtin_amdgcn_wmma_f32_16x16x32_bf16(
                    false, af.vv, false, bf.vv, (short)0, acc[st],
                    false, false);
            }
        }

        // ---- Fused epilogue for this g-tile: tanh(acc + hb[g]) * v[g],
        //      then reduce over the 16 n-columns (xor shuffles) and merge
        //      across waves/g-tiles via ds_add_f32 into sLog. Doing this
        //      per-tile keeps no long-lived partial registers alive. ----
        const float vg  = v[g];
        const float hbg = hb[b * GG + g];
#pragma unroll
        for (int st = 0; st < STILES; ++st)
#pragma unroll
            for (int r = 0; r < 8; ++r) {
                float x = TANHF(acc[st][r] + hbg) * vg;
                x += __shfl_xor(x, 1, 32);
                x += __shfl_xor(x, 2, 32);
                x += __shfl_xor(x, 4, 32);
                x += __shfl_xor(x, 8, 32);
                if (lr == 0) {
                    // lane 0 holds rows m=r, lane 16 holds rows m=8+r
                    atomicAdd(&sLog[st * 16 + r + 8 * hh], x);
                }
            }
    }
    __syncthreads();

    if (t < 64) logits[(size_t)b * SS + s0 + t] = sLog[t];
}

// ---------------------------------------------------------------------------
// Kernel 3: row softmax over S=2048 per batch. grid = 16, block = 256.
// ---------------------------------------------------------------------------
__global__ void softmax_kernel(const float* __restrict__ logits,
                               float* __restrict__ out) {
    const int b = blockIdx.x;
    const int t = threadIdx.x;
    __shared__ float sred[8];
    const float* row = logits + (size_t)b * SS;

    float lv[8];
    float mx = -INFINITY;
#pragma unroll
    for (int i = 0; i < 8; ++i) {
        lv[i] = row[t + i * 256];
        mx = fmaxf(mx, lv[i]);
    }
#pragma unroll
    for (int m = 16; m >= 1; m >>= 1) mx = fmaxf(mx, __shfl_xor(mx, m, 32));
    if ((t & 31) == 0) sred[t >> 5] = mx;
    __syncthreads();
    mx = sred[0];
#pragma unroll
    for (int i = 1; i < 8; ++i) mx = fmaxf(mx, sred[i]);
    __syncthreads();

    float s = 0.0f;
#pragma unroll
    for (int i = 0; i < 8; ++i) {
        lv[i] = __expf(lv[i] - mx);
        s += lv[i];
    }
#pragma unroll
    for (int m = 16; m >= 1; m >>= 1) s += __shfl_xor(s, m, 32);
    if ((t & 31) == 0) sred[t >> 5] = s;
    __syncthreads();
    s = 0.0f;
#pragma unroll
    for (int i = 0; i < 8; ++i) s += sred[i];
    const float inv = 1.0f / s;
#pragma unroll
    for (int i = 0; i < 8; ++i)
        out[(size_t)b * SS + t + i * 256] = lv[i] * inv;
}

// ---------------------------------------------------------------------------
extern "C" void kernel_launch(void* const* d_in, const int* in_sizes, int n_in,
                              void* d_out, int out_size, void* d_ws,
                              size_t ws_size, hipStream_t stream) {
    (void)in_sizes; (void)n_in; (void)out_size; (void)ws_size;
    const float* hidden = (const float*)d_in[0];   // (16, 1024)
    const float* enc    = (const float*)d_in[1];   // (16, 2048, 1024)
    const float* W      = (const float*)d_in[2];   // (1024, 2048)
    const float* bias   = (const float*)d_in[3];   // (1024,)
    const float* v      = (const float*)d_in[4];   // (1024,)
    float* out = (float*)d_out;                    // (16, 1, 2048) f32

    float* hb     = (float*)d_ws;                  // 16*1024 f32 (64 KB)
    float* logits = hb + BB * GG;                  // 16*2048 f32 (128 KB)

    prep_hb_kernel<<<dim3(GG / 16), dim3(256), 0, stream>>>(hidden, W, bias, hb);
    attn_logits_kernel<<<dim3(SS / (STILES * 16), BB), dim3(256), SMEM_BYTES,
                         stream>>>(enc, W, hb, v, logits);
    softmax_kernel<<<dim3(BB), dim3(256), 0, stream>>>(logits, out);
}
